// GatedDeltaNet_12171937317591
// MI455X (gfx1250) — matrix operand order repaired and checked by reading, not compiled, not run on hardware
//
#include <hip/hip_runtime.h>
#include <math.h>

typedef __attribute__((ext_vector_type(16))) __bf16 v16bf;
typedef __attribute__((ext_vector_type(8)))  __bf16 v8bf;
typedef __attribute__((ext_vector_type(4)))  __bf16 v4bf;
typedef __attribute__((ext_vector_type(8)))  float  v8f;

#define NHEADS 12
#define DKH    64
#define DVH    128
#define KCONV  4
#define BSZ    4
#define LSEQ   2048
#define HDIM   1024
#define KDIM   768
#define VDIM   1536
#define ROWS   (BSZ * LSEQ)   // 8192

__device__ __forceinline__ float sigmoidf_(float x) { return 1.0f / (1.0f + expf(-x)); }
__device__ __forceinline__ float siluf_(float x) { return x * sigmoidf_(x); }

__device__ __forceinline__ v4bf cvt4(float4 f) {
  v4bf r;
  r[0] = (__bf16)f.x; r[1] = (__bf16)f.y; r[2] = (__bf16)f.z; r[3] = (__bf16)f.w;
  return r;
}

// Per-thread staging registers for one 64x64x64 K-chunk.
struct StageRegs {
  float4 ta[4];       // A row segment
  float  tb[4][4];    // B 4x4 micro-block
};

__device__ __forceinline__ void stage_load(const float* __restrict__ aptr,
                                           const float* __restrict__ bptr,
                                           int N, StageRegs& r) {
#pragma unroll
  for (int i = 0; i < 4; ++i)
    r.ta[i] = *reinterpret_cast<const float4*>(aptr + i * 16);
#pragma unroll
  for (int j = 0; j < 4; ++j)
    *reinterpret_cast<float4*>(r.tb[j]) =
        *reinterpret_cast<const float4*>(bptr + (size_t)j * N);
}

__device__ __forceinline__ void stage_store(__bf16 (*As)[72], __bf16 (*Bs)[72],
                                            int arow, int asub, int br, int bn,
                                            const StageRegs& r) {
#pragma unroll
  for (int i = 0; i < 4; ++i)
    *reinterpret_cast<v4bf*>(&As[arow][asub + i * 16]) = cvt4(r.ta[i]);
#pragma unroll
  for (int i = 0; i < 4; ++i) {
    v4bf col;
    col[0] = (__bf16)r.tb[0][i]; col[1] = (__bf16)r.tb[1][i];
    col[2] = (__bf16)r.tb[2][i]; col[3] = (__bf16)r.tb[3][i];
    *reinterpret_cast<v4bf*>(&Bs[bn + i][br]) = col;
  }
}

// A fragment (16x32 bf16): lane<16 -> K 0..15,
// lane>=16 -> K 16..31; row M = lane%16.
__device__ __forceinline__ v16bf load_afrag(const __bf16* ap, bool hiLane) {
  v8bf a0 = *reinterpret_cast<const v8bf*>(ap + (hiLane ? 16 : 0));
  v8bf a1 = *reinterpret_cast<const v8bf*>(ap + 8 + (hiLane ? 16 : 0));
  return __builtin_shufflevector(a0, a1, 0,1,2,3,4,5,6,7,8,9,10,11,12,13,14,15);
}

// B fragment (32x16 bf16): lane<16 -> K 0..15, lane>=16 -> K 16..31; N = lane%16.
__device__ __forceinline__ v16bf load_bfrag(const __bf16* bp) {
  v8bf b0 = *reinterpret_cast<const v8bf*>(bp);
  v8bf b1 = *reinterpret_cast<const v8bf*>(bp + 8);
  return __builtin_shufflevector(b0, b1, 0,1,2,3,4,5,6,7,8,9,10,11,12,13,14,15);
}

// 2 K-subchunks x 2 N-tiles = 4 wmma. All 12 ds_load_b128 issued up front
// (in-order DS returns -> partial dscnt waits), then 4 back-to-back wmmas.
__device__ __forceinline__ void compute_tile(const __bf16 (*As)[72],
                                             const __bf16 (*Bs)[72],
                                             int mrow, int ncol0, bool hiLane,
                                             v8f& acc0, v8f& acc1) {
  const int bko = hiLane ? 16 : 0;
  v16bf af0  = load_afrag(&As[mrow][0],  hiLane);
  v16bf bf00 = load_bfrag(&Bs[ncol0][bko]);
  v16bf bf01 = load_bfrag(&Bs[ncol0 + 16][bko]);
  v16bf af1  = load_afrag(&As[mrow][32], hiLane);
  v16bf bf10 = load_bfrag(&Bs[ncol0][32 + bko]);
  v16bf bf11 = load_bfrag(&Bs[ncol0 + 16][32 + bko]);

  acc0 = __builtin_amdgcn_wmma_f32_16x16x32_bf16(
      false, af0, false, bf00, (short)0, acc0, false, false);
  acc1 = __builtin_amdgcn_wmma_f32_16x16x32_bf16(
      false, af0, false, bf01, (short)0, acc1, false, false);
  acc0 = __builtin_amdgcn_wmma_f32_16x16x32_bf16(
      false, af1, false, bf10, (short)0, acc0, false, false);
  acc1 = __builtin_amdgcn_wmma_f32_16x16x32_bf16(
      false, af1, false, bf11, (short)0, acc1, false, false);
}

// ---------------------------------------------------------------------------
// Tiled bf16 WMMA GEMM: C[M,N] = A[M,K] * B[K,N], fp32 in/out, bf16 compute.
// Block tile 64(M) x 64(N), K-step 64, double-buffered LDS software pipeline
// with sched_barrier fences so global loads for chunk c are issued BEFORE the
// WMMAs on chunk c-1 (HBM latency hidden behind matrix math), then chunk c is
// converted+staged; one barrier per chunk.
// 256 threads = 8 waves; wave w: M-stripe (w&3)*16, N-stripes (w>>2)*32,+16.
// Requires M%64==0, N%64==0, K%64==0 (true for all shapes used here).
// ---------------------------------------------------------------------------
__global__ __launch_bounds__(256) void gemm_wmma_bf16(
    const float* __restrict__ A, const float* __restrict__ B,
    float* __restrict__ C, int M, int N, int K)
{
  // Row stride 72 bf16 = 144B (multiple of 16B) -> aligned b64 stores / b128 loads.
  __shared__ __align__(16) __bf16 As[2][64][72];   // [buf][m][k]
  __shared__ __align__(16) __bf16 Bs[2][64][72];   // [buf][n][k] (transposed)

  const int tid  = threadIdx.x;
  const int wave = tid >> 5;
  const int lane = tid & 31;
  const bool hiLane = (lane >= 16);
  const int m0 = blockIdx.y * 64;
  const int n0 = blockIdx.x * 64;
  const int wm = (wave & 3) * 16;        // 0..48
  const int wn = (wave >> 2) * 32;       // 0 / 32

  // A staging: thread (row = tid>>2, sub = tid&3) loads 4 float4 from one row.
  const int arow = tid >> 2;
  const int asub = (tid & 3) * 4;
  const float* aptr = A + (size_t)(m0 + arow) * K + asub;
  // B staging: thread owns a 4x4 micro-block: br = (tid>>4)*4, bn = (tid&15)*4.
  const int br = (tid >> 4) * 4;
  const int bn = (tid & 15) * 4;
  const float* bptr = B + (size_t)br * N + (n0 + bn);
  const size_t bstep = (size_t)64 * N;

  v8f acc0 = {0.f,0.f,0.f,0.f,0.f,0.f,0.f,0.f};
  v8f acc1 = {0.f,0.f,0.f,0.f,0.f,0.f,0.f,0.f};

  const int mrow  = wm + (lane & 15);
  const int ncol0 = wn + (lane & 15);
  const int nchunks = K >> 6;

  // Prologue: stage chunk 0.
  StageRegs r;
  stage_load(aptr, bptr, N, r);
  __builtin_prefetch(aptr + 64, 0, 1);        // -> global_prefetch_b8
  __builtin_prefetch(bptr + bstep, 0, 1);
  stage_store(As[0], Bs[0], arow, asub, br, bn, r);
  __syncthreads();

  for (int c = 1; c < nchunks; ++c) {
    aptr += 64;
    bptr += bstep;
    // Issue global loads for chunk c ...
    stage_load(aptr, bptr, N, r);
    if (c + 1 < nchunks) {                    // prefetch chunk c+1
      __builtin_prefetch(aptr + 64, 0, 1);
      __builtin_prefetch(bptr + bstep, 0, 1);
    }
    // Fence: loads stay above, WMMAs below -> load latency overlaps math.
    __builtin_amdgcn_sched_barrier(0);
    // ... WMMAs on chunk c-1 while chunk-c loads are in flight ...
    compute_tile(As[(c - 1) & 1], Bs[(c - 1) & 1], mrow, ncol0, hiLane, acc0, acc1);
    // Fence: keep the loadcnt-wait / cvt / ds_store phase after the WMMAs.
    __builtin_amdgcn_sched_barrier(0);
    // ... then convert + stage chunk c (buffer c&1 last read in iter c-1,
    // separated by that iteration's barrier).
    stage_store(As[c & 1], Bs[c & 1], arow, asub, br, bn, r);
    __syncthreads();
  }
  compute_tile(As[(nchunks - 1) & 1], Bs[(nchunks - 1) & 1],
               mrow, ncol0, hiLane, acc0, acc1);

  // C/D layout: VGPR r -> M = r + (lane>=16 ? 8 : 0), N = lane%16.
  const int ccol  = n0 + wn + (lane & 15);
  const int mbase = m0 + wm + (hiLane ? 8 : 0);
#pragma unroll
  for (int rr = 0; rr < 8; ++rr) {
    float* crow = C + (size_t)(mbase + rr) * N;
    crow[ccol]      = acc0[rr];
    crow[ccol + 16] = acc1[rr];
  }
}

// ---------------------------------------------------------------------------
// Small-N projections: gk = -exp(A_log)*softplus(x@Wgk + dt_bias),
// beta = sigmoid(x@Wb + bb).  One block (768 thr = 24 waves) per row; the
// activation row is staged in LDS once; wave w computes output w
// (0..11 -> gk head, 12..23 -> beta head) with a shuffle reduction.
// ---------------------------------------------------------------------------
__global__ __launch_bounds__(768) void proj_small_kernel(
    const float* __restrict__ X, const float* __restrict__ Wgk,
    const float* __restrict__ Wb, const float* __restrict__ bb,
    const float* __restrict__ A_log, const float* __restrict__ dt_bias,
    float* __restrict__ gk, float* __restrict__ beta)
{
  const int row  = blockIdx.x;
  const int tid  = threadIdx.x;
  const int wave = tid >> 5;            // 0..23
  const int lane = tid & 31;

  __shared__ float sx[HDIM];
  for (int c = tid; c < HDIM; c += 768)
    sx[c] = X[(size_t)row * HDIM + c];
  __syncthreads();

  const int h = wave % NHEADS;
  const float* W = (wave < NHEADS) ? Wgk : Wb;

  float s = 0.f;
  for (int c = lane; c < HDIM; c += 32)
    s += sx[c] * W[(size_t)c * NHEADS + h];
#pragma unroll
  for (int off = 16; off > 0; off >>= 1)
    s += __shfl_down(s, off, 32);

  if (lane == 0) {
    if (wave < NHEADS) {
      float x = s + dt_bias[h];
      float sp = (x > 20.f) ? x : log1pf(expf(x));   // softplus
      gk[(size_t)row * NHEADS + h] = -expf(A_log[h]) * sp;
    } else {
      beta[(size_t)row * NHEADS + h] = sigmoidf_(s + bb[h]);
    }
  }
}

// ---------------------------------------------------------------------------
// Causal depthwise conv (K=4) + silu + per-head L2 norm for q / k.
// Block = 64 threads = one (b,l,h); out layout [B,L,NH,DK].
// ---------------------------------------------------------------------------
__global__ __launch_bounds__(64) void conv_qk_kernel(
    const float* __restrict__ pre, const float* __restrict__ w,
    float* __restrict__ out)
{
  const int bl = blockIdx.x;            // b*L + l
  const int h  = blockIdx.y;
  const int b  = bl / LSEQ, l = bl % LSEQ;
  const int cl = threadIdx.x;           // 0..63
  const int c  = h * DKH + cl;

  float acc = 0.f;
#pragma unroll
  for (int j = 0; j < KCONV; ++j) {
    int ls = l - (KCONV - 1) + j;
    if (ls >= 0)
      acc += pre[((size_t)b * LSEQ + ls) * KDIM + c] * w[c * KCONV + j];
  }
  float x = siluf_(acc);

  __shared__ float ss[64];
  ss[cl] = x * x;
  __syncthreads();
#pragma unroll
  for (int off = 32; off > 0; off >>= 1) {
    if (cl < off) ss[cl] += ss[cl + off];
    __syncthreads();
  }
  float n = fmaxf(sqrtf(ss[0]), 1e-12f);
  out[(((size_t)b * LSEQ + l) * NHEADS + h) * DKH + cl] = x / n;
}

// ---------------------------------------------------------------------------
// Causal depthwise conv (K=4) + silu for v. One thread per element.
// ---------------------------------------------------------------------------
__global__ __launch_bounds__(256) void conv_v_kernel(
    const float* __restrict__ pre, const float* __restrict__ w,
    float* __restrict__ out)
{
  size_t idx = (size_t)blockIdx.x * 256 + threadIdx.x;   // over B*L*VDIM
  int c = (int)(idx % VDIM);
  size_t bl = idx / VDIM;
  int l = (int)(bl % LSEQ);
  size_t brow = bl - (size_t)l;   // b*LSEQ

  float acc = 0.f;
#pragma unroll
  for (int j = 0; j < KCONV; ++j) {
    int ls = l - (KCONV - 1) + j;
    if (ls >= 0)
      acc += pre[(brow + ls) * VDIM + c] * w[c * KCONV + j];
  }
  out[idx] = siluf_(acc);
}

// ---------------------------------------------------------------------------
// Gated delta-rule recurrence. Columns of S are independent:
//   S[:,v] = e^g * S[:,v] + k * (beta*(v_v - e^g * k.S_old[:,v]));  o_v = q.S[:,v]
// One wave (32 threads) per (b, h, 32-column v-tile) -> 192 single-wave
// workgroups (workgroup barriers are S_NOP). S[64] kept in VGPRs per thread.
// ---------------------------------------------------------------------------
__global__ __launch_bounds__(32) void recurrence_kernel(
    const float* __restrict__ q, const float* __restrict__ k,
    const float* __restrict__ v, const float* __restrict__ gk,
    const float* __restrict__ beta, float* __restrict__ o)
{
  const int vb = blockIdx.x & 3;                 // v tile (DV/32 = 4)
  const int h  = (blockIdx.x >> 2) % NHEADS;
  const int b  = blockIdx.x / (4 * NHEADS);
  const int v0 = vb * 32;
  const int lane = threadIdx.x;

  __shared__ float sk[DKH], sq[DKH], sv[32];
  __shared__ float sgb[2];

  float S[DKH];
#pragma unroll
  for (int i = 0; i < DKH; ++i) S[i] = 0.f;

  for (int l = 0; l < LSEQ; ++l) {
    const size_t base_qk = (((size_t)b * LSEQ + l) * NHEADS + h) * (size_t)DKH;
    const size_t base_v  = (((size_t)b * LSEQ + l) * NHEADS + h) * (size_t)DVH;
    sk[lane]      = k[base_qk + lane];
    sk[lane + 32] = k[base_qk + lane + 32];
    sq[lane]      = q[base_qk + lane];
    sq[lane + 32] = q[base_qk + lane + 32];
    sv[lane]      = v[base_v + v0 + lane];
    if (lane == 0) {
      sgb[0] = gk[((size_t)b * LSEQ + l) * NHEADS + h];
      sgb[1] = beta[((size_t)b * LSEQ + l) * NHEADS + h];
    }
    __syncthreads();

    const float decay = expf(sgb[0]);
    float rk[DKH];
    float pred = 0.f;
#pragma unroll
    for (int i = 0; i < DKH; ++i) {
      rk[i] = sk[i];
      pred += rk[i] * S[i];
    }
    const float delta = sgb[1] * (sv[lane] - decay * pred);

    float oacc = 0.f;
#pragma unroll
    for (int i = 0; i < DKH; ++i) {
      float sn = S[i] * decay + rk[i] * delta;
      S[i] = sn;
      oacc += sq[i] * sn;
    }
    o[base_v + v0 + lane] = oacc;
    __syncthreads();
  }
}

// ---------------------------------------------------------------------------
// Fused RMSNorm(o) * gnorm_w * silu(g). Block = 128 threads = one (row, head).
// ---------------------------------------------------------------------------
__global__ __launch_bounds__(128) void norm_gate_kernel(
    const float* __restrict__ o, const float* __restrict__ g,
    const float* __restrict__ gnorm_w, float* __restrict__ y)
{
  const int row = blockIdx.x;
  const int h   = blockIdx.y;
  const int vt  = threadIdx.x;   // 0..127
  const size_t base = ((size_t)row * NHEADS + h) * (size_t)DVH;

  float ov = o[base + vt];
  __shared__ float ss[DVH];
  ss[vt] = ov * ov;
  __syncthreads();
#pragma unroll
  for (int off = 64; off > 0; off >>= 1) {
    if (vt < off) ss[vt] += ss[vt + off];
    __syncthreads();
  }
  float rms = rsqrtf(ss[0] / (float)DVH + 1e-5f);
  float gv = g[base + vt];
  y[base + vt] = ov * rms * gnorm_w[vt] * siluf_(gv);
}

// ---------------------------------------------------------------------------
extern "C" void kernel_launch(void* const* d_in, const int* in_sizes, int n_in,
                              void* d_out, int out_size, void* d_ws, size_t ws_size,
                              hipStream_t stream)
{
  (void)in_sizes; (void)n_in; (void)out_size; (void)ws_size;

  const float* hs    = (const float*)d_in[0];   // [B,L,H]
  const float* Wq    = (const float*)d_in[1];   // [H,Kd]
  const float* Wk    = (const float*)d_in[2];   // [H,Kd]
  const float* Wv    = (const float*)d_in[3];   // [H,Vd]
  const float* Wg    = (const float*)d_in[4];   // [H,Vd]
  const float* Wgk   = (const float*)d_in[5];   // [H,NH]
  const float* Wb    = (const float*)d_in[6];   // [H,NH]
  const float* bb    = (const float*)d_in[7];   // [NH]
  const float* convq = (const float*)d_in[8];   // [Kd,KC]
  const float* convk = (const float*)d_in[9];   // [Kd,KC]
  const float* convv = (const float*)d_in[10];  // [Vd,KC]
  const float* A_log = (const float*)d_in[11];  // [NH]
  const float* gnw   = (const float*)d_in[12];  // [DV]
  const float* Wo    = (const float*)d_in[13];  // [Vd,H]
  const float* dtb   = (const float*)d_in[14];  // [NH]
  float* out = (float*)d_out;                   // [B,L,H]

  const size_t R = ROWS;
  float* ws    = (float*)d_ws;
  float* q_pre = ws;                         // R*768
  float* k_pre = q_pre + R * KDIM;           // R*768
  float* v_pre = k_pre + R * KDIM;           // R*1536
  float* gbuf  = v_pre + R * VDIM;           // R*1536
  float* qn    = gbuf  + R * VDIM;           // R*768
  float* kn    = qn    + R * KDIM;           // R*768
  float* vs    = kn    + R * KDIM;           // R*1536
  float* gkb   = vs    + R * VDIM;           // R*12
  float* betab = gkb   + R * NHEADS;         // R*12
  float* obuf  = q_pre;   // reuse q_pre+k_pre (contiguous R*1536) after convs
  float* ybuf  = v_pre;   // reuse v_pre after conv_v

  const dim3 blk256(256);

  // Projections (bf16 WMMA GEMMs, 64x64x64 tiles, double-buffered pipeline)
  gemm_wmma_bf16<<<dim3(KDIM / 64, ROWS / 64), blk256, 0, stream>>>(hs, Wq, q_pre, ROWS, KDIM, HDIM);
  gemm_wmma_bf16<<<dim3(KDIM / 64, ROWS / 64), blk256, 0, stream>>>(hs, Wk, k_pre, ROWS, KDIM, HDIM);
  gemm_wmma_bf16<<<dim3(VDIM / 64, ROWS / 64), blk256, 0, stream>>>(hs, Wv, v_pre, ROWS, VDIM, HDIM);
  gemm_wmma_bf16<<<dim3(VDIM / 64, ROWS / 64), blk256, 0, stream>>>(hs, Wg, gbuf, ROWS, VDIM, HDIM);

  // gk / beta (N=12 each): one block per row, 24 waves
  proj_small_kernel<<<dim3(ROWS), dim3(768), 0, stream>>>(
      hs, Wgk, Wb, bb, A_log, dtb, gkb, betab);

  // Causal depthwise conv + silu (+ l2norm for q,k)
  conv_qk_kernel<<<dim3(ROWS, NHEADS), dim3(64), 0, stream>>>(q_pre, convq, qn);
  conv_qk_kernel<<<dim3(ROWS, NHEADS), dim3(64), 0, stream>>>(k_pre, convk, kn);
  conv_v_kernel<<<dim3((ROWS * VDIM) / 256), dim3(256), 0, stream>>>(v_pre, convv, vs);

  // Gated delta-rule recurrence: 4*12*4 = 192 single-wave workgroups
  recurrence_kernel<<<dim3(BSZ * NHEADS * (DVH / 32)), dim3(32), 0, stream>>>(
      qn, kn, vs, gkb, betab, obuf);

  // RMSNorm * w * silu(gate)
  norm_gate_kernel<<<dim3(ROWS, NHEADS), dim3(128), 0, stream>>>(obuf, gbuf, gnw, ybuf);

  // Output projection (bf16 WMMA GEMM)
  gemm_wmma_bf16<<<dim3(HDIM / 64, ROWS / 64), blk256, 0, stream>>>(ybuf, Wo, out, ROWS, HDIM, VDIM);
}
